// VectorQuantizer_76416058130938
// MI455X (gfx1250) — matrix operand order repaired and checked by reading, not compiled, not run on hardware
//
#include <hip/hip_runtime.h>
#include <hip/hip_bf16.h>

// ---------------------------------------------------------------------------
// VQ-VAE vector quantizer for MI455X (gfx1250, wave32, WMMA).
// z: [16, 512, 64, 64] f32, codebook: [1024, 512] f32.
// Outputs (flat in d_out): z_q [33554432] f32, loss [1] f32, indices [65536] f32.
// ---------------------------------------------------------------------------

typedef __attribute__((ext_vector_type(16))) __bf16 v16bf;
typedef __attribute__((ext_vector_type(8)))  float  v8f;

union BF16x16 {
    uint4 q[2];
    v16bf v;
};

#define N_PIX      65536        // 16*64*64
#define E_DIM      512
#define N_E        1024
#define HWSZ       4096         // 64*64
#define LOSS_OFF   33554432
#define IDX_OFF    33554433
#define M_TILE     128          // pixels per vq_argmin workgroup (8 waves)
#define G_TILE     64           // pixels per vq_gather workgroup
#define LDS_STRIDE 520          // 512 + 8 bf16 pad (keeps 16B align, spreads banks)

__device__ __forceinline__ unsigned short f2bf(float f) {
    unsigned u = __builtin_bit_cast(unsigned, f);
    unsigned r = u + 0x7FFFu + ((u >> 16) & 1u);   // round-to-nearest-even
    return (unsigned short)(r >> 16);
}

// --------------------------------------------------------------------------
// Kernel 1: per-entry squared norms of the codebook (f32).
// --------------------------------------------------------------------------
__global__ void vq_prep_norm(const float* __restrict__ cb, float* __restrict__ norm) {
    int n = blockIdx.x * blockDim.x + threadIdx.x;
    if (n < N_E) {
        const float4* row = (const float4*)(cb + (size_t)n * E_DIM);
        float s = 0.f;
        for (int k = 0; k < E_DIM / 4; ++k) {
            float4 v = row[k];
            s += v.x * v.x + v.y * v.y + v.z * v.z + v.w * v.w;
        }
        norm[n] = s;
    }
}

// --------------------------------------------------------------------------
// Kernel 2: codebook f32 -> bf16, pre-swizzled into WMMA B-fragment order.
// B (32x16, K x N) k order chosen here (contiguous; A uses the same):
//   lane l: column N = l%16; lanes 0-15 hold K=0..15, lanes 16-31 K=16..31,
//   VGPR v packs K=2v,2v+1. So each lane needs 16 *consecutive* K values:
//   a contiguous 32-byte chunk of the (row-major) codebook row.
// Fragment array: frag[((t*16 + s)*32 + lane)] of 32 B, t = n-tile, s = k-step.
// One thread per (t, s, lane): 32768 threads total.
// --------------------------------------------------------------------------
__global__ void vq_prep_cbfrag(const float* __restrict__ cb,
                               unsigned short* __restrict__ frag) {
    int g = blockIdx.x * blockDim.x + threadIdx.x;       // 0 .. 32767
    int t = g >> 9;                                       // n-tile   0..63
    int s = (g >> 5) & 15;                                // k-step   0..15
    int l = g & 31;                                       // lane     0..31
    int n  = t * 16 + (l & 15);
    int kb = s * 32 + ((l & 16) ? 16 : 0);
    const float* src = cb + (size_t)n * E_DIM + kb;
    unsigned w[8];
#pragma unroll
    for (int i = 0; i < 8; ++i) {
        unsigned short lo = f2bf(src[2 * i]);
        unsigned short hi = f2bf(src[2 * i + 1]);
        w[i] = (unsigned)lo | ((unsigned)hi << 16);
    }
    uint4* dst = (uint4*)frag + (size_t)g * 2;
    dst[0] = make_uint4(w[0], w[1], w[2], w[3]);
    dst[1] = make_uint4(w[4], w[5], w[6], w[7]);
}

// --------------------------------------------------------------------------
// Kernel 3: WMMA distance + argmin.
// Block = 256 threads (8 waves), handles 128 consecutive pixels.
// Wave w owns the 16-row m-tile m0 = 16*w; its A fragments (full K=512,
// 16 x v16bf = 128 VGPRs) stay register-resident across all 64 n-tiles.
// Each 16 KB codebook-fragment tile is staged into LDS ONCE per block
// (double-buffered, one barrier per tile) and consumed by all 8 waves:
// exactly one L2 read of the fragment stream per block.
// dist(n) = ||e_n||^2 - 2 * dot(z, e_n)   (||z||^2 dropped: row-constant).
// --------------------------------------------------------------------------
__global__ __launch_bounds__(256) void vq_argmin(
        const float* __restrict__ z,
        const unsigned short* __restrict__ frag,
        const float* __restrict__ norm,
        float* __restrict__ idxOut) {
    __shared__ __align__(16) unsigned short shz[M_TILE * LDS_STRIDE];   // 133,120 B
    __shared__ __align__(16) unsigned short shb[2][8192];               //  32,768 B
    __shared__ float shn[N_E];                                          //   4,096 B

    const int tid = threadIdx.x;
    const int n0  = blockIdx.x * M_TILE;                  // first pixel
    const int b   = n0 >> 12;
    const int hw  = n0 & (HWSZ - 1);
    const size_t zbase = (size_t)b * (E_DIM * HWSZ) + hw;

    // ---- stage z tile into LDS as bf16 (coalesced float4 along pixels) ----
    {
        const int coff = tid >> 5;          // 0..7 channels per pass
        const int m4   = (tid & 31) * 4;    // 4 consecutive pixels (0..124)
        for (int cc = 0; cc < E_DIM; cc += 8) {
            const int c = cc + coff;
            const float4 v = *(const float4*)(z + zbase + (size_t)c * HWSZ + m4);
            shz[(m4 + 0) * LDS_STRIDE + c] = f2bf(v.x);
            shz[(m4 + 1) * LDS_STRIDE + c] = f2bf(v.y);
            shz[(m4 + 2) * LDS_STRIDE + c] = f2bf(v.z);
            shz[(m4 + 3) * LDS_STRIDE + c] = f2bf(v.w);
        }
    }
    for (int i = tid; i < N_E; i += 256) shn[i] = norm[i];

    const uint4* fq = (const uint4*)frag;     // fragment stream, 1024 uint4 per tile
    uint4* sb0 = (uint4*)shb[0];
    uint4* sb1 = (uint4*)shb[1];

    // ---- stage B tile 0 (256 threads x 64 B, perfectly coalesced) ----
#pragma unroll
    for (int i = 0; i < 4; ++i) sb0[tid * 4 + i] = fq[tid * 4 + i];
    __syncthreads();

    const int lane  = tid & 31;
    const int m0    = (tid >> 5) * 16;
    const int row   = m0 + (lane & 15);
    const int khalf = (lane & 16) ? 16 : 0;  // A: lanes0-15 K 0..15, lanes16-31 K 16..31

    // ---- A fragments, register-resident for the whole n loop ----
    v16bf a[16];
#pragma unroll
    for (int s = 0; s < 16; ++s) {
        BF16x16 u;
        u.q[0] = *(const uint4*)&shz[row * LDS_STRIDE + s * 32 + khalf];
        u.q[1] = *(const uint4*)&shz[row * LDS_STRIDE + s * 32 + khalf + 8];
        a[s] = u.v;
    }

    float best[8];
    int   bidx[8];
#pragma unroll
    for (int j = 0; j < 8; ++j) { best[j] = 3.4e38f; bidx[j] = 0; }

    for (int t = 0; t < 64; ++t) {
        uint4* bb = (t & 1) ? sb1 : sb0;
        uint4* bn = (t & 1) ? sb0 : sb1;

        // issue global loads for the next tile (overlap with WMMA chain)
        uint4 r[4];
        if (t < 63) {
            const int gbase = (t + 1) * 1024 + tid * 4;
#pragma unroll
            for (int i = 0; i < 4; ++i) r[i] = fq[gbase + i];
        }

        // 16-step K chain: B from LDS (32 B/lane contiguous -> conflict-free)
        v8f acc = {};
#pragma unroll
        for (int s = 0; s < 16; ++s) {
            const int ci = (s * 32 + lane) * 2;
            BF16x16 u;
            u.q[0] = bb[ci];
            u.q[1] = bb[ci + 1];
            acc = __builtin_amdgcn_wmma_f32_16x16x32_bf16(
                      false, a[s], false, u.v, (short)0, acc, false, false);
        }

        // D layout: VGPR j <-> row M=j (lanes 0-15) or M=8+j (lanes 16-31); N = lane%16
        const int   n  = t * 16 + (lane & 15);
        const float nv = shn[n];
#pragma unroll
        for (int j = 0; j < 8; ++j) {
            const float dist = nv - 2.0f * acc[j];
            if (dist < best[j]) { best[j] = dist; bidx[j] = n; }   // strict '<' keeps first min
        }

        // commit next tile to the other buffer, then one barrier per tile
        if (t < 63) {
#pragma unroll
            for (int i = 0; i < 4; ++i) bn[tid * 4 + i] = r[i];
        }
        __syncthreads();
    }

    // ---- argmin reduction across the 16-lane N groups ----
#pragma unroll
    for (int off = 1; off < 16; off <<= 1) {
#pragma unroll
        for (int j = 0; j < 8; ++j) {
            const float ob = __shfl_xor(best[j], off, 32);
            const int   oi = __shfl_xor(bidx[j], off, 32);
            if (ob < best[j] || (ob == best[j] && oi < bidx[j])) {
                best[j] = ob; bidx[j] = oi;
            }
        }
    }
    if ((lane & 15) == 0) {
        const int mrow = m0 + ((lane & 16) ? 8 : 0);
#pragma unroll
        for (int j = 0; j < 8; ++j)
            idxOut[n0 + mrow + j] = (float)bidx[j];
    }
}

// --------------------------------------------------------------------------
// Kernel 4: gather z_q = codebook[idx], write (same [B,C,H,W] addressing as z),
// and accumulate squared error per block (deterministic partials).
// --------------------------------------------------------------------------
__global__ __launch_bounds__(256) void vq_gather(
        const float* __restrict__ z,
        const float* __restrict__ cb,
        const float* __restrict__ idxF,
        float* __restrict__ out,
        float* __restrict__ partial) {
    __shared__ int   sidx[G_TILE];
    __shared__ float red[256];

    const int tid = threadIdx.x;
    const int n0  = blockIdx.x * G_TILE;
    const int b   = n0 >> 12;
    const int hw  = n0 & (HWSZ - 1);
    const size_t base = (size_t)b * (E_DIM * HWSZ) + hw;

    if (tid < G_TILE) sidx[tid] = (int)idxF[n0 + tid];
    __syncthreads();

    const int m    = tid & 63;
    const int coff = tid >> 6;                 // 4 channels per pass
    const int crow = sidx[m];
    float sse = 0.f;
    for (int cc = 0; cc < E_DIM; cc += 4) {
        const int c = cc + coff;
        const size_t off = base + (size_t)c * HWSZ + m;
        const float zv = z[off];
        const float q  = cb[(size_t)crow * E_DIM + c];   // L2-resident gather
        out[off] = q;
        const float d = q - zv;
        sse += d * d;
    }
    red[tid] = sse;
    __syncthreads();
    for (int sft = 128; sft > 0; sft >>= 1) {
        if (tid < sft) red[tid] += red[tid + sft];
        __syncthreads();
    }
    if (tid == 0) partial[blockIdx.x] = red[0];
}

// --------------------------------------------------------------------------
// Kernel 5: deterministic final reduction of 1024 partials -> scalar loss.
// codebook_loss = (1 + BETA) * mean((z_q - z)^2), BETA = 0.25.
// --------------------------------------------------------------------------
__global__ void vq_final(const float* __restrict__ partial, float* __restrict__ lossOut) {
    __shared__ float red[256];
    const int tid = threadIdx.x;
    red[tid] = partial[tid] + partial[tid + 256] + partial[tid + 512] + partial[tid + 768];
    __syncthreads();
    for (int sft = 128; sft > 0; sft >>= 1) {
        if (tid < sft) red[tid] += red[tid + sft];
        __syncthreads();
    }
    if (tid == 0) *lossOut = red[0] * (1.25f / 33554432.0f);
}

// --------------------------------------------------------------------------
extern "C" void kernel_launch(void* const* d_in, const int* in_sizes, int n_in,
                              void* d_out, int out_size, void* d_ws, size_t ws_size,
                              hipStream_t stream) {
    const float* z  = (const float*)d_in[0];
    const float* cb = (const float*)d_in[1];
    float* out     = (float*)d_out;
    float* lossOut = out + LOSS_OFF;
    float* idxOut  = out + IDX_OFF;

    // workspace layout
    unsigned short* frag = (unsigned short*)d_ws;                         // 1 MB  (bf16 B-fragments)
    float* norm    = (float*)((char*)d_ws + (1 << 20));                   // 4 KB  (||e||^2)
    float* partial = (float*)((char*)d_ws + (1 << 20) + 4096);            // 4 KB  (block partials)

    vq_prep_norm  <<<  4, 256, 0, stream>>>(cb, norm);
    vq_prep_cbfrag<<<128, 256, 0, stream>>>(cb, frag);
    vq_argmin     <<<N_PIX / M_TILE, 256, 0, stream>>>(z, frag, norm, idxOut);
    vq_gather     <<<N_PIX / G_TILE, 256, 0, stream>>>(z, cb, idxOut, out, partial);
    vq_final      <<<  1, 256, 0, stream>>>(partial, lossOut);
}
